// BloomAttention_55327768707800
// MI455X (gfx1250) — matrix operand order repaired and checked by reading, not compiled, not run on hardware
//
#include <hip/hip_runtime.h>

// ---------------------------------------------------------------------------
// Types for CDNA5 WMMA (wave32): V_WMMA_F32_16X16X32_BF16
// ---------------------------------------------------------------------------
typedef __bf16 bf16_t;
typedef __attribute__((ext_vector_type(16))) __bf16 v16bf;
typedef __attribute__((ext_vector_type(8)))  __bf16 v8bf;
typedef __attribute__((ext_vector_type(4)))  __bf16 v4bf;
typedef __attribute__((ext_vector_type(8)))  float  v8f;
typedef __attribute__((ext_vector_type(4)))  unsigned int v4u;
typedef __attribute__((ext_vector_type(8)))  int v8i;
typedef __attribute__((ext_vector_type(4)))  int v4i;

#define S_LEN 2048
#define HID   2048
#define NHEAD 16
#define HD    128

static __device__ __forceinline__ v8f vzero8() {
  v8f r;
#pragma unroll
  for (int i = 0; i < 8; ++i) r[i] = 0.0f;
  return r;
}

// CDNA5 async global->LDS copy (bypasses VGPRs, tracked by ASYNCcnt).
static __device__ __forceinline__ void async_ld16(const bf16_t* lds_ptr,
                                                  const bf16_t* gptr) {
  const unsigned int lds_off = (unsigned int)(unsigned long long)(const void*)lds_ptr;
  asm volatile("global_load_async_to_lds_b128 %0, %1, off"
               :: "v"(lds_off), "v"(gptr)
               : "memory");
}
static __device__ __forceinline__ void wait_async0() {
  asm volatile("s_wait_asynccnt 0x0" ::: "memory");
}

// ---------------------------------------------------------------------------
// Tensor Data Mover: 2D tile (tile_d0 x tile_d1, 2-byte elements) from a
// row-major tensor (row stride = stride_d0 elements) into LDS, with hardware
// row padding: 4 DWORDs (16B) inserted after every 16 DWORDs (64B), i.e. an
// 80-byte LDS row stride for a 32-element bf16 tile row.
// D# packing per CDNA5 ISA §8.3 (group0) / §8.4 (group1); groups 2,3 unused.
// ---------------------------------------------------------------------------
static __device__ __forceinline__ void tdm_load_2d(const bf16_t* lds_ptr,
                                                   const bf16_t* gptr,
                                                   unsigned tensor_d0,
                                                   unsigned tensor_d1,
                                                   unsigned tile_d0,
                                                   unsigned tile_d1,
                                                   unsigned stride_d0) {
  const unsigned lds_addr = __builtin_amdgcn_readfirstlane(
      (int)(unsigned int)(unsigned long long)(const void*)lds_ptr);
  const unsigned long long ga = (unsigned long long)gptr;
  const unsigned ga_lo = __builtin_amdgcn_readfirstlane((int)(unsigned)(ga & 0xffffffffull));
  const unsigned ga_hi = __builtin_amdgcn_readfirstlane((int)(unsigned)(ga >> 32));

  v4u g0;
  g0[0] = 1u;                                        // count=1, user descriptor
  g0[1] = lds_addr;                                  // LDS byte address
  g0[2] = ga_lo;                                     // global_addr[31:0]
  g0[3] = (ga_hi & 0x01ffffffu) | (2u << 30);        // global_addr[56:32], type=2

  v8i g1;
  g1[0] = (int)((1u << 16)        // data_size = 2 bytes
                | (1u << 20)      // pad_enable
                | (3u << 22)      // pad_interval: 16 DWORDs
                | (3u << 25));    // pad_amount:   4 DWORDs
  g1[1] = (int)((tensor_d0 & 0xffffu) << 16);                    // dim0[15:0]
  g1[2] = (int)(((tensor_d0 >> 16) & 0xffffu) |
                ((tensor_d1 & 0xffffu) << 16));                  // dim0 hi, dim1 lo
  g1[3] = (int)(((tensor_d1 >> 16) & 0xffffu) |
                ((tile_d0 & 0xffffu) << 16));                    // dim1 hi, tile_dim0
  g1[4] = (int)(tile_d1 & 0xffffu);                              // tile_dim1 (tile_dim2=0)
  g1[5] = (int)stride_d0;                                        // dim0_stride[31:0]
  g1[6] = 0;                                                     // stride hi, dim1_stride lo
  g1[7] = 0;

  const v4i z4 = {0, 0, 0, 0};
  const v8i z8 = {0, 0, 0, 0, 0, 0, 0, 0};
  __builtin_amdgcn_tensor_load_to_lds(g0, g1, z4, z4, z8, 0);
}

// A-matrix frag (16x32 bf16, ISA 7.12.2): lane row = lane%16, half g=lane/16
// holds K = {g*8..g*8+7} and {16+g*8..16+g*8+7}.
static __device__ __forceinline__ v16bf ld_a(const bf16_t* row, int g) {
  const v8bf lo = *(const v8bf*)(row + 8 * g);
  const v8bf hi = *(const v8bf*)(row + 16 + 8 * g);
  v16bf r;
#pragma unroll
  for (int i = 0; i < 8; ++i) { r[i] = lo[i]; r[i + 8] = hi[i]; }
  return r;
}

// B-matrix frag (32x16 bf16): lane col = lane%16, half g holds K = {g*8..g*8+7} and {16+g*8..16+g*8+7}.
static __device__ __forceinline__ v16bf ld_b(const bf16_t* col, int g) {
  const v8bf lo = *(const v8bf*)(col + 8 * g);
  const v8bf hi = *(const v8bf*)(col + 16 + 8 * g);
  v16bf r;
#pragma unroll
  for (int i = 0; i < 8; ++i) { r[i] = lo[i]; r[i + 8] = hi[i]; }
  return r;
}

static __device__ __forceinline__ v8f wmma_bf16(v16bf a, v16bf b, v8f c) {
  return __builtin_amdgcn_wmma_f32_16x16x32_bf16(
      false, a, false, b, (short)0, c, false, false);
}

// ---------------------------------------------------------------------------
// fp32 -> bf16 bulk convert (4 elements / thread)
// ---------------------------------------------------------------------------
__global__ __launch_bounds__(256) void cvt_kernel(const float* __restrict__ in,
                                                  bf16_t* __restrict__ out, int n4) {
  const int i = blockIdx.x * blockDim.x + threadIdx.x;
  if (i >= n4) return;
  const float4 v = reinterpret_cast<const float4*>(in)[i];
  v4bf o;
  o[0] = (bf16_t)v.x; o[1] = (bf16_t)v.y; o[2] = (bf16_t)v.z; o[3] = (bf16_t)v.w;
  reinterpret_cast<v4bf*>(out)[i] = o;
}

// ---------------------------------------------------------------------------
// GEMM core: workgroup tile 64(M) x 256(N), 8 waves (2x4), wave tile 32x64,
// K-step 32, double-buffered LDS filled by the Tensor Data Mover (wave 0
// issues one DMA per tile; TENSORcnt + barrier gates the consumers).
// A [M,K] row-major bf16, W [N,K] row-major bf16 (NT GEMM, K contiguous both).
// ---------------------------------------------------------------------------
#define GEMM_BODY(A_, W_, NB_, C_)                                             \
  __shared__ __align__(16) bf16_t sA[2][64 * 40];                              \
  __shared__ __align__(16) bf16_t sB[2][256 * 40];                             \
  const int tid = threadIdx.x;                                                 \
  const int wave = tid >> 5, lane = tid & 31;                                  \
  const int g = lane >> 4, l16 = lane & 15;                                    \
  const int wm = wave >> 2, wn = wave & 3;                                     \
  const int rowBase = blockIdx.y * 64;                                         \
  const int colBase = blockIdx.x * 256;                                        \
  v8f C_[2][4];                                                                \
  _Pragma("unroll") for (int i = 0; i < 2; ++i)                                \
      _Pragma("unroll") for (int j = 0; j < 4; ++j) C_[i][j] = vzero8();       \
  if (wave == 0) { /* prologue: DMA tile 0 into buffer 0 */                    \
    tdm_load_2d(sA[0], A_ + (size_t)rowBase * HID, HID, 2048, 32, 64, HID);    \
    tdm_load_2d(sB[0], W_ + (size_t)colBase * HID, HID, NB_, 32, 256, HID);    \
  }                                                                            \
  for (int k0 = 0; k0 < HID; k0 += 32) {                                       \
    const int cur = (k0 >> 5) & 1;                                             \
    if (wave == 0) __builtin_amdgcn_s_wait_tensorcnt(0); /* buf(cur) ready */  \
    __syncthreads();                                                           \
    if (k0 + 32 < HID && wave == 0) { /* overlap next DMA with compute */      \
      tdm_load_2d(sA[cur ^ 1], A_ + (size_t)rowBase * HID + k0 + 32,           \
                  HID, 2048, 32, 64, HID);                                     \
      tdm_load_2d(sB[cur ^ 1], W_ + (size_t)colBase * HID + k0 + 32,           \
                  HID, NB_, 32, 256, HID);                                     \
    }                                                                          \
    const bf16_t* pa = sA[cur];                                                \
    const bf16_t* pb = sB[cur];                                                \
    const v16bf a0 = ld_a(pa + (wm * 32 + l16) * 40, g);                       \
    const v16bf a1 = ld_a(pa + (wm * 32 + 16 + l16) * 40, g);                  \
    v16bf bfr[4];                                                              \
    _Pragma("unroll") for (int ni = 0; ni < 4; ++ni)                           \
        bfr[ni] = ld_b(pb + (wn * 64 + ni * 16 + l16) * 40, g);                \
    _Pragma("unroll") for (int ni = 0; ni < 4; ++ni) {                         \
      C_[0][ni] = wmma_bf16(a0, bfr[ni], C_[0][ni]);                           \
      C_[1][ni] = wmma_bf16(a1, bfr[ni], C_[1][ni]);                           \
    }                                                                          \
  }

// ---------------------------------------------------------------------------
// QKV GEMM: epilogue adds bias, folds 1/sqrt(HD) into Q, scatters to
// per-head [NH,S,HD] bf16 Q/K/V buffers.
// ---------------------------------------------------------------------------
__global__ __launch_bounds__(256) void gemm_qkv_kernel(
    const bf16_t* __restrict__ A,     // [2048,2048] hidden bf16
    const bf16_t* __restrict__ W,     // [6144,2048] W_qkv bf16
    const float*  __restrict__ bias,  // [6144]
    bf16_t* __restrict__ qb, bf16_t* __restrict__ kb, bf16_t* __restrict__ vb) {
  GEMM_BODY(A, W, 6144, c)
  const float inv_norm = 0.08838834764831845f;  // 1/sqrt(128)
#pragma unroll
  for (int mi = 0; mi < 2; ++mi)
#pragma unroll
    for (int ni = 0; ni < 4; ++ni) {
      const int n = colBase + wn * 64 + ni * 16 + l16;
      const int headi = n / 384;  // qkv cols packed as [NH, 3, HD]
      const int which = (n % 384) / 128;
      const int d = n % 128;
      const float bv = bias[n];
      bf16_t* dst = (which == 0) ? qb : (which == 1) ? kb : vb;
      const float scale = (which == 0) ? inv_norm : 1.0f;
#pragma unroll
      for (int r = 0; r < 8; ++r) {  // C layout: VGPR r -> row r + 8*half
        const int row = rowBase + wm * 32 + mi * 16 + r + 8 * g;
        const float val = (c[mi][ni][r] + bv) * scale;
        dst[(size_t)headi * S_LEN * HD + (size_t)row * HD + d] = (bf16_t)val;
      }
    }
}

// ---------------------------------------------------------------------------
// Dense GEMM: out = ctx @ W_dense^T + b_dense + residual (fp32 out)
// ---------------------------------------------------------------------------
__global__ __launch_bounds__(256) void gemm_dense_kernel(
    const bf16_t* __restrict__ A,         // [2048,2048] ctx bf16
    const bf16_t* __restrict__ W,         // [2048,2048] W_dense bf16
    const float*  __restrict__ bias,      // [2048]
    const float*  __restrict__ residual,  // [2048,2048]
    float* __restrict__ out) {
  GEMM_BODY(A, W, 2048, c)
#pragma unroll
  for (int mi = 0; mi < 2; ++mi)
#pragma unroll
    for (int ni = 0; ni < 4; ++ni) {
      const int n = colBase + wn * 64 + ni * 16 + l16;
      const float bv = bias[n];
#pragma unroll
      for (int r = 0; r < 8; ++r) {
        const int row = rowBase + wm * 32 + mi * 16 + r + 8 * g;
        out[(size_t)row * HID + n] =
            c[mi][ni][r] + bv + residual[(size_t)row * HID + n];
      }
    }
}

// ---------------------------------------------------------------------------
// Flash attention: grid (S/128, NH). 8 waves, each owns 16 q rows.
// K tile staged via async-to-LDS DMA (overlaps the V transpose staging);
// online softmax; P re-laid to A-operand layout via padded per-wave LDS.
// ---------------------------------------------------------------------------
__global__ __launch_bounds__(256) void attn_kernel(
    const bf16_t* __restrict__ qb, const bf16_t* __restrict__ kb,
    const bf16_t* __restrict__ vb, const float* __restrict__ alibi,
    bf16_t* __restrict__ ctx /* [S, H] bf16, heads merged */) {
  __shared__ __align__(16) bf16_t sK[64 * 136];    // kv-row major, stride 272B
  __shared__ __align__(16) bf16_t sVt[128 * 72];   // d-major (transposed)
  __shared__ __align__(16) bf16_t sP[8 * 16 * 72]; // per-wave P
  const int tid = threadIdx.x;
  const int wave = tid >> 5, lane = tid & 31;
  const int g = lane >> 4, l16 = lane & 15;
  const int head = blockIdx.y;
  const int qTile = blockIdx.x * 128 + wave * 16;

  // q tile (16 x 128) as 4 A-frags straight from global (rows K-contiguous)
  const bf16_t* qrow = qb + ((size_t)head * S_LEN + qTile + l16) * HD;
  v16bf qf[4];
#pragma unroll
  for (int f = 0; f < 4; ++f) qf[f] = ld_a(qrow + 32 * f, g);

  v8f o[8];
#pragma unroll
  for (int h = 0; h < 8; ++h) o[h] = vzero8();
  float mrow[8], lrow[8];
#pragma unroll
  for (int r = 0; r < 8; ++r) { mrow[r] = -3.0e38f; lrow[r] = 0.0f; }
  bf16_t* sPw = sP + wave * 16 * 72;

  const int kvEnd = blockIdx.x * 128 + 128;  // causal: skip all-future tiles
  for (int kv0 = 0; kv0 < kvEnd; kv0 += 64) {
    // K tile via async DMA (no VGPR round-trip) ...
#pragma unroll
    for (int i = 0; i < 2; ++i) {
      const int idx = tid + i * 256;
      const int r = idx >> 3, ch = idx & 7;
      async_ld16(sK + r * 136 + ch * 8,
                 kb + ((size_t)head * S_LEN + kv0 + r) * HD + ch * 8);
    }
    // ... overlapped with V transpose through registers
    v8bf vv[2];
#pragma unroll
    for (int i = 0; i < 2; ++i) {
      const int idx = tid + i * 256;
      const int r = idx >> 3, ch = idx & 7;
      vv[i] = *(const v8bf*)(vb + ((size_t)head * S_LEN + kv0 + r) * HD + ch * 8);
    }
#pragma unroll
    for (int i = 0; i < 2; ++i) {
      const int idx = tid + i * 256;
      const int r = idx >> 3, ch = idx & 7;
#pragma unroll
      for (int j = 0; j < 8; ++j) sVt[(ch * 8 + j) * 72 + r] = vv[i][j];
    }
    wait_async0();
    __syncthreads();

    // scores S = q @ k^T : 4 n-chunks x 4 K-steps of 32
    v8f sc[4];
#pragma unroll
    for (int a = 0; a < 4; ++a) {
      sc[a] = vzero8();
#pragma unroll
      for (int f = 0; f < 4; ++f) {
        const v16bf bk = ld_b(sK + (a * 16 + l16) * 136 + 32 * f, g);
        sc[a] = wmma_bf16(qf[f], bk, sc[a]);
      }
    }
    // alibi + causal mask (q already carries 1/sqrt(HD))
#pragma unroll
    for (int a = 0; a < 4; ++a) {
      const int kvIdx = kv0 + a * 16 + l16;
      const float al = alibi[head * S_LEN + kvIdx];
#pragma unroll
      for (int r = 0; r < 8; ++r) {
        const int row = qTile + r + 8 * g;
        float v = sc[a][r] + al;
        if (kvIdx > row) v = -1.0e30f;
        sc[a][r] = v;
      }
    }
    // online softmax: row max across 16 lanes (xor 8,4,2,1 stays in half-wave)
    float corr[8];
#pragma unroll
    for (int r = 0; r < 8; ++r) {
      float rm = fmaxf(fmaxf(sc[0][r], sc[1][r]), fmaxf(sc[2][r], sc[3][r]));
#pragma unroll
      for (int off = 8; off >= 1; off >>= 1)
        rm = fmaxf(rm, __shfl_xor(rm, off, 32));
      const float mnew = fmaxf(mrow[r], rm);
      corr[r] = __expf(mrow[r] - mnew);
      mrow[r] = mnew;
    }
#pragma unroll
    for (int h = 0; h < 8; ++h)
#pragma unroll
      for (int r = 0; r < 8; ++r) o[h][r] *= corr[r];

    float rsum[8];
#pragma unroll
    for (int r = 0; r < 8; ++r) rsum[r] = 0.0f;
#pragma unroll
    for (int a = 0; a < 4; ++a) {
#pragma unroll
      for (int r = 0; r < 8; ++r) {
        const float p = __expf(sc[a][r] - mrow[r]);
        rsum[r] += p;
        sPw[(r + 8 * g) * 72 + a * 16 + l16] = (bf16_t)p;  // C-layout -> LDS
      }
    }
#pragma unroll
    for (int r = 0; r < 8; ++r) {
#pragma unroll
      for (int off = 8; off >= 1; off >>= 1)
        rsum[r] += __shfl_xor(rsum[r], off, 32);
      lrow[r] = lrow[r] * corr[r] + rsum[r];
    }
    __syncthreads();

    // O += P(16x64) @ V(64x128)
    const v16bf pf0 = ld_a(sPw + l16 * 72, g);
    const v16bf pf1 = ld_a(sPw + l16 * 72 + 32, g);
#pragma unroll
    for (int h = 0; h < 8; ++h) {
      const v16bf bv0 = ld_b(sVt + (h * 16 + l16) * 72, g);
      const v16bf bv1 = ld_b(sVt + (h * 16 + l16) * 72 + 32, g);
      o[h] = wmma_bf16(pf0, bv0, o[h]);
      o[h] = wmma_bf16(pf1, bv1, o[h]);
    }
    __syncthreads();
  }

  // finalize: O / l, merge heads into [S, H] bf16
#pragma unroll
  for (int r = 0; r < 8; ++r) {
    const float inv = 1.0f / lrow[r];
    const int row = qTile + r + 8 * g;
#pragma unroll
    for (int h = 0; h < 8; ++h)
      ctx[(size_t)row * HID + head * HD + h * 16 + l16] = (bf16_t)(o[h][r] * inv);
  }
}

// ---------------------------------------------------------------------------
// kernel_launch
// ---------------------------------------------------------------------------
extern "C" void kernel_launch(void* const* d_in, const int* in_sizes, int n_in,
                              void* d_out, int out_size, void* d_ws, size_t ws_size,
                              hipStream_t stream) {
  (void)in_sizes; (void)n_in; (void)out_size; (void)ws_size;
  const float* hidden   = (const float*)d_in[0];
  const float* residual = (const float*)d_in[1];
  const float* alibi    = (const float*)d_in[2];
  // d_in[3] = attention_mask (bool causal) — computed analytically in-kernel
  const float* W_qkv    = (const float*)d_in[4];
  const float* b_qkv    = (const float*)d_in[5];
  const float* W_dense  = (const float*)d_in[6];
  const float* b_dense  = (const float*)d_in[7];

  char* ws = (char*)d_ws;
  bf16_t* hbf   = (bf16_t*)(ws);                       //  8 MB  hidden bf16
  bf16_t* wqbf  = (bf16_t*)(ws + ((size_t)8  << 20));  // 24 MB  W_qkv bf16
  bf16_t* wdbf  = (bf16_t*)(ws + ((size_t)32 << 20));  //  8 MB  W_dense bf16
  bf16_t* qbf   = (bf16_t*)(ws + ((size_t)40 << 20));  //  8 MB  Q [NH,S,HD]
  bf16_t* kbf   = (bf16_t*)(ws + ((size_t)48 << 20));  //  8 MB  K
  bf16_t* vbf   = (bf16_t*)(ws + ((size_t)56 << 20));  //  8 MB  V
  bf16_t* ctxbf = (bf16_t*)(ws + ((size_t)64 << 20));  //  8 MB  ctx [S,H]

  cvt_kernel<<<4096, 256, 0, stream>>>(hidden, hbf, 1048576);
  cvt_kernel<<<12288, 256, 0, stream>>>(W_qkv, wqbf, 3145728);
  cvt_kernel<<<4096, 256, 0, stream>>>(W_dense, wdbf, 1048576);

  gemm_qkv_kernel<<<dim3(24, 32), 256, 0, stream>>>(hbf, wqbf, b_qkv, qbf, kbf, vbf);
  attn_kernel<<<dim3(16, 16), 256, 0, stream>>>(qbf, kbf, vbf, alibi, ctxbf);
  gemm_dense_kernel<<<dim3(8, 32), 256, 0, stream>>>(ctxbf, wdbf, b_dense, residual,
                                                     (float*)d_out);
}
